// AttackRMultiHeadAttention_82248623719016
// MI455X (gfx1250) — matrix operand order repaired and checked by reading, not compiled, not run on hardware
//
#include <hip/hip_runtime.h>
#include <hip/hip_bf16.h>

typedef __attribute__((ext_vector_type(16))) _Float16 v16h;
typedef __attribute__((ext_vector_type(8)))  _Float16 v8h;
typedef __attribute__((ext_vector_type(8)))  float    v8f;
typedef __attribute__((ext_vector_type(4)))  float    v4f;

#define ROWP 1032   // padded LDS row (1032 % 64 == 8 -> avoids bank conflicts)

// ---- output layout (floats) ----
// mixed_q  [2,1024,512]       @ 0
// mixed_k  [2,1024,512]       @ 1048576
// vh       [2,8,1024,64]      @ 2097152
// probs    [2,8,1024,1024]    @ 3145728
// origprob [2,8,1024,1024]    @ 19922944

__device__ __forceinline__ v16h cat8(v8h lo, v8h hi) {
    return __builtin_shufflevector(lo, hi, 0,1,2,3,4,5,6,7,8,9,10,11,12,13,14,15);
}

// ---------------- prep: f32 -> f16 input copy ----------------
__global__ void cvt_x_kernel(const float* __restrict__ x, _Float16* __restrict__ xh, int n) {
    int i = blockIdx.x * blockDim.x + threadIdx.x;
    if (i < n) xh[i] = (_Float16)x[i];
}

// ---------------- prep: transpose weights to f16 [out,in] ----------------
__global__ void transpose_w_kernel(const float* __restrict__ Wq, const float* __restrict__ Wk,
                                   const float* __restrict__ Wv, _Float16* __restrict__ wt) {
    int idx = blockIdx.x * blockDim.x + threadIdx.x;  // 0..262143
    int z = blockIdx.y;
    const float* W = (z == 0) ? Wq : (z == 1) ? Wk : Wv;
    int k = idx >> 9, n = idx & 511;
    wt[(size_t)z * 262144 + (size_t)n * 512 + k] = (_Float16)W[(size_t)k * 512 + n];
}

// ---------------- QKV projection: one wave per 16x32 tile (2 N-tiles, shared A) ----------------
__global__ void qkv_wmma_kernel(const _Float16* __restrict__ xh,
                                const _Float16* __restrict__ wt,
                                const float* __restrict__ bq, const float* __restrict__ bk,
                                const float* __restrict__ bv,
                                float* __restrict__ out,
                                _Float16* __restrict__ qh, _Float16* __restrict__ kh) {
    const int lane  = threadIdx.x;            // 0..31
    const int mrow0 = blockIdx.x * 16;        // 0..2032
    const int ncol0 = blockIdx.y * 32;        // 0..480
    const int z     = blockIdx.z;             // 0=Q 1=K 2=V
    const int r16 = lane & 15, half = lane >> 4;

    const float* bias = (z == 0) ? bq : (z == 1) ? bk : bv;
    const _Float16* wz    = wt + (size_t)z * (512 * 512);
    const _Float16* arow  = xh + (size_t)(mrow0 + r16) * 512;
    const _Float16* bcol0 = wz + (size_t)(ncol0 + r16) * 512;
    const _Float16* bcol1 = wz + (size_t)(ncol0 + 16 + r16) * 512;

    float bv0 = bias[ncol0 + r16];
    float bv1 = bias[ncol0 + 16 + r16];
    v8f acc0, acc1;
#pragma unroll
    for (int r = 0; r < 8; ++r) { acc0[r] = bv0; acc1[r] = bv1; }

#pragma unroll 4
    for (int k = 0; k < 512; k += 32) {
        v16h a  = *(const v16h*)(arow + k + 16 * half);
        v16h b0 = *(const v16h*)(bcol0 + k + 16 * half);
        v16h b1 = *(const v16h*)(bcol1 + k + 16 * half);
        acc0 = __builtin_amdgcn_wmma_f32_16x16x32_f16(false, a, false, b0, (short)0, acc0, false, false);
        acc1 = __builtin_amdgcn_wmma_f32_16x16x32_f16(false, a, false, b1, (short)0, acc1, false, false);
    }

#pragma unroll
    for (int r = 0; r < 8; ++r) {
        int m = mrow0 + r + 8 * half;
        int bb = m >> 10, s = m & 1023;
#pragma unroll
        for (int u = 0; u < 2; ++u) {
            int n = ncol0 + u * 16 + r16;
            float v = (u == 0) ? acc0[r] : acc1[r];
            int h = n >> 6, d = n & 63;
            size_t hidx = ((size_t)((bb * 8 + h) * 1024 + s)) * 64 + d;   // heads layout
            if (z == 0)      { out[(size_t)m * 512 + n] = v;            qh[hidx] = (_Float16)v; }
            else if (z == 1) { out[1048576 + (size_t)m * 512 + n] = v;  kh[hidx] = (_Float16)v; }
            else             { out[2097152 + hidx] = v; }
        }
    }
}

// ---------------- per-row affine factors: q_ord/q_dst/k_ord/k_dst ----------------
__global__ void rowfactor_kernel(const float* __restrict__ out,
                                 const float* __restrict__ ow, const float* __restrict__ dw,
                                 float* __restrict__ qord, float* __restrict__ qdst,
                                 float* __restrict__ kord, float* __restrict__ kdst) {
    int idx = blockIdx.x * blockDim.x + threadIdx.x;  // bh*1024 + i, 0..16383
    int bh = idx >> 10, i = idx & 1023;
    int b = bh >> 3, h = bh & 7;
    const float* qrow = out + ((size_t)(b * 1024 + i)) * 512 + h * 64;
    const float* krow = out + 1048576 + ((size_t)(b * 1024 + i)) * 512 + h * 64;
    float qo = 0.f, qd = 0.f, ko = 0.f, kd = 0.f;
#pragma unroll 8
    for (int d = 0; d < 64; ++d) {
        qo += qrow[d] * ow[d];      qd += qrow[d] * dw[d];
        ko += krow[d] * ow[64 + d]; kd += krow[d] * dw[64 + d];
    }
    qord[idx] = qo; qdst[idx] = qd; kord[idx] = ko; kdst[idx] = kd;
}

// ---------------- fused scores + biases + dual softmax ----------------
// grid (16, 64): (b*H+h, i_tile). 256 threads = 8 waves. LDS: 16x1024 strip x2 + tables.
__global__ void attn_fused_kernel(const _Float16* __restrict__ qh, const _Float16* __restrict__ kh,
                                  const float* __restrict__ mask,
                                  const float* __restrict__ qord, const float* __restrict__ qdst,
                                  const float* __restrict__ kord, const float* __restrict__ kdst,
                                  const float* __restrict__ obp, const float* __restrict__ dbp,
                                  const float* __restrict__ scp,
                                  float* __restrict__ out) {
    extern __shared__ float lds[];
    float* sb  = lds;               // [16][ROWP] scores -> masked orig logits
    float* ab  = lds + 16 * ROWP;   // [16][ROWP] masked adjusted logits
    float* koL = lds + 32 * ROWP;   // [1024]
    float* kdL = koL + 1024;        // [1024]
    float* qoL = kdL + 1024;        // [16]
    float* qdL = qoL + 16;          // [16]
    float* gdL = qdL + 16;          // [1024] log(|i-j|+1) table

    const int tid = threadIdx.x;
    const int wave = tid >> 5, lane = tid & 31;
    const int bh = blockIdx.x, it = blockIdx.y;
    const int r16 = lane & 15, half = lane >> 4;

    // phase 0: stage column factors (b128) + distance table + row factors
    {
        int j4 = tid * 4;  // 0..1020, covers 1024 exactly
        *(v4f*)(koL + j4) = *(const v4f*)(kord + bh * 1024 + j4);
        *(v4f*)(kdL + j4) = *(const v4f*)(kdst + bh * 1024 + j4);
    }
    for (int t = tid; t < 1024; t += 256)
        gdL[t] = __logf((float)t + 1.f);
    if (tid < 16) {
        qoL[tid] = qord[bh * 1024 + it * 16 + tid];
        qdL[tid] = qdst[bh * 1024 + it * 16 + tid];
    }

    // phase 1: scores Q[16,64] x K^T[64,1024] via WMMA; wave w handles j-tiles w, w+8, ...
    const _Float16* qrow = qh + ((size_t)(bh * 1024 + it * 16 + r16)) * 64;
    v16h a0 = *(const v16h*)(qrow + 16 * half);
    v16h a1 = *(const v16h*)(qrow + 32 + 16 * half);

    for (int t = 0; t < 8; ++t) {
        int jt = wave + t * 8;
        const _Float16* krow = kh + ((size_t)(bh * 1024 + jt * 16 + r16)) * 64;
        v16h b0 = *(const v16h*)(krow + 16 * half);
        v16h b1 = *(const v16h*)(krow + 32 + 16 * half);
        v8f acc = {};
        acc = __builtin_amdgcn_wmma_f32_16x16x32_f16(false, a0, false, b0, (short)0, acc, false, false);
        acc = __builtin_amdgcn_wmma_f32_16x16x32_f16(false, a1, false, b1, (short)0, acc, false, false);
#pragma unroll
        for (int r = 0; r < 8; ++r)
            sb[(r + 8 * half) * ROWP + jt * 16 + r16] = acc[r];
    }
    __syncthreads();

    // phase 2: biases + dual softmax; wave w owns rows {w, w+8}; all traffic in float4
    const float order_b = obp[0], dist_b = dbp[0];
    const float sc2 = scp[0] * scp[0];
    const float invscale = 0.125f;   // 1/sqrt(64)
    const int b = bh >> 3;

    for (int rr = 0; rr < 2; ++rr) {
        int rloc = wave + rr * 8;
        int ig = it * 16 + rloc;
        const float* mrow = mask + (size_t)b * 1048576 + (size_t)ig * 1024;
        __builtin_prefetch(mrow + lane * 32, 0, 1);      // global_prefetch_b8: warm mask row
        float qo = qoL[rloc], qd = qdL[rloc];
        float* srow = sb + rloc * ROWP;
        float* arow = ab + rloc * ROWP;
        float maxa = -3.4e38f, maxo = -3.4e38f;

        for (int j = lane * 4; j < 1024; j += 128) {
            v4f sc4 = *(const v4f*)(srow + j);
            v4f ko4 = *(const v4f*)(koL + j);
            v4f kd4 = *(const v4f*)(kdL + j);
            v4f m4  = *(const v4f*)(mrow + j);
            v4f av, ov;
#pragma unroll
            for (int e = 0; e < 4; ++e) {
                int jj = j + e;
                float pr = 1.f / (1.f + __expf(-(qo + ko4[e] + order_b)));
                float eo = (jj > ig) ? __logf(pr + 1e-24f) : __logf(1.f - pr + 1e-24f);
                int ad = jj - ig; ad = (ad < 0) ? -ad : ad;
                float gd = gdL[ad];
                float pd = qd + kd4[e] + dist_b;
                float ed = -0.5f * (gd - pd) * (gd - pd) * sc2;
                av[e] = (sc4[e] + eo + ed) * invscale + m4[e];
                ov[e] = sc4[e] * invscale + m4[e];
                maxa = fmaxf(maxa, av[e]); maxo = fmaxf(maxo, ov[e]);
            }
            *(v4f*)(arow + j) = av;
            *(v4f*)(srow + j) = ov;
        }
#pragma unroll
        for (int off = 16; off > 0; off >>= 1) {
            maxa = fmaxf(maxa, __shfl_xor(maxa, off, 32));
            maxo = fmaxf(maxo, __shfl_xor(maxo, off, 32));
        }

        float suma = 0.f, sumo = 0.f;
        for (int j = lane * 4; j < 1024; j += 128) {
            v4f a4 = *(const v4f*)(arow + j);
            v4f o4 = *(const v4f*)(srow + j);
#pragma unroll
            for (int e = 0; e < 4; ++e) {
                suma += __expf(a4[e] - maxa);
                sumo += __expf(o4[e] - maxo);
            }
        }
#pragma unroll
        for (int off = 16; off > 0; off >>= 1) {
            suma += __shfl_xor(suma, off, 32);
            sumo += __shfl_xor(sumo, off, 32);
        }
        float ra = 1.f / suma, ro = 1.f / sumo;

        float* pp = out + 3145728  + ((size_t)(bh * 1024 + ig)) * 1024;
        float* pq = out + 19922944 + ((size_t)(bh * 1024 + ig)) * 1024;
        for (int j = lane * 4; j < 1024; j += 128) {
            v4f a4 = *(const v4f*)(arow + j);
            v4f o4 = *(const v4f*)(srow + j);
            v4f p4, q4;
#pragma unroll
            for (int e = 0; e < 4; ++e) {
                p4[e] = __expf(a4[e] - maxa) * ra;
                q4[e] = __expf(o4[e] - maxo) * ro;
            }
            *(v4f*)(pp + j) = p4;   // b128 stores: the HBM-bound stream
            *(v4f*)(pq + j) = q4;
        }
    }
}

extern "C" void kernel_launch(void* const* d_in, const int* in_sizes, int n_in,
                              void* d_out, int out_size, void* d_ws, size_t ws_size,
                              hipStream_t stream) {
    const float* x   = (const float*)d_in[0];
    const float* msk = (const float*)d_in[1];
    const float* Wq  = (const float*)d_in[2];
    const float* bq  = (const float*)d_in[3];
    const float* Wk  = (const float*)d_in[4];
    const float* bk  = (const float*)d_in[5];
    const float* Wv  = (const float*)d_in[6];
    const float* bv  = (const float*)d_in[7];
    const float* ow  = (const float*)d_in[8];
    const float* ob  = (const float*)d_in[9];
    const float* dw  = (const float*)d_in[10];
    const float* db  = (const float*)d_in[11];
    const float* sc  = (const float*)d_in[12];
    float* out = (float*)d_out;
    char* ws = (char*)d_ws;

    // workspace layout (byte offsets)
    _Float16* xh = (_Float16*)(ws);                       // 2048*512 f16      (2 MB)
    _Float16* wt = (_Float16*)(ws + (size_t)2097152);     // 3*512*512 f16     (1.5 MB)
    _Float16* qh = (_Float16*)(ws + (size_t)3670016);     // heads f16         (2 MB)
    _Float16* kh = (_Float16*)(ws + (size_t)5767168);     // heads f16         (2 MB)
    float* qord  = (float*)(ws + (size_t)7864320);        // 16384 f32 each
    float* qdst  = qord + 16384;
    float* kord  = qdst + 16384;
    float* kdst  = kord + 16384;

    cvt_x_kernel<<<4096, 256, 0, stream>>>(x, xh, 1048576);
    transpose_w_kernel<<<dim3(1024, 3), 256, 0, stream>>>(Wq, Wk, Wv, wt);
    qkv_wmma_kernel<<<dim3(128, 16, 3), 32, 0, stream>>>(xh, wt, bq, bk, bv, out, qh, kh);
    rowfactor_kernel<<<64, 256, 0, stream>>>(out, ow, dw, qord, qdst, kord, kdst);

    size_t shmem = (size_t)(2 * 16 * ROWP + 2 * 1024 + 32 + 1024) * sizeof(float);  // 144,512 B
    attn_fused_kernel<<<dim3(16, 64), 256, shmem, stream>>>(
        qh, kh, msk, qord, qdst, kord, kdst, ob, db, sc, out);
}
